// MLP_85727547228464
// MI455X (gfx1250) — matrix operand order repaired and checked by reading, not compiled, not run on hardware
//
#include <hip/hip_runtime.h>

// ---------------------------------------------------------------------------
// SNN MLP block for MI455X (gfx1250, wave32, WMMA + Tensor Data Mover).
//   x:[T,B,C,HW] -> LIF -> (1x1 conv+BN, C->Ch) -> LIF -> (1x1 conv+BN, Ch->C)
// T=8 B=32 C=384 Ch=1536 HW=256.
//
// - GEMMs in f16 WMMA (v_wmma_f32_16x16x32_f16, f32 accum). Spikes are {0,1}
//   (lossless in f16); only weights are rounded f32->f16.
// - GEMM1 keeps all 8 time steps in 8 v8f accumulators per wave so the second
//   LIF scan runs entirely in registers (no f32 hidden round-trip).
// - All GEMM tile movement uses TDM (tensor_load_to_lds/tensor_store_from_lds)
//   with double-buffered LDS: the DMA for k-step n+1 overlaps compute of step
//   n; one barrier per step; zero VGPR staging. TENSORcnt synchronizes.
// - Workspace: ~255 MB (spikes1 50MB f16, spikes2 201MB f16, weights).
// ---------------------------------------------------------------------------

#define T_  8
#define B_  32
#define C_  384
#define CH_ 1536
#define HW_ 256
#define INV_TAU (1.0f / 1.5f)

typedef __attribute__((ext_vector_type(16))) _Float16 v16h;
typedef __attribute__((ext_vector_type(8)))  _Float16 v8h;
typedef __attribute__((ext_vector_type(8)))  float    v8f;
typedef __attribute__((ext_vector_type(4)))  unsigned int u32x4;
typedef __attribute__((ext_vector_type(8)))  int i32x8;
typedef __attribute__((ext_vector_type(4)))  int i32x4;

union HFrag { v16h v; v8h h[2]; };

// ---------------------------------------------------------------------------
// Tensor Data Mover support (gfx1250). Descriptor per CDNA5 ISA §8 (D#).
// ---------------------------------------------------------------------------
#if defined(__has_builtin)
#if __has_builtin(__builtin_amdgcn_tensor_load_to_lds) && \
    __has_builtin(__builtin_amdgcn_tensor_store_from_lds) && \
    __has_builtin(__builtin_amdgcn_s_wait_tensorcnt)
#define USE_TDM 1
#endif
#endif
#ifndef USE_TDM
#define USE_TDM 0
#endif

struct TdmDesc { u32x4 g0; i32x8 g1; i32x4 g2; i32x4 g3; };

// Builds a D# for an up-to-3D tile whose tensor dims equal the tile dims
// (global_addr points at the tile start, so no OOB clipping is wanted).
//   data_size_log2: 0=1B,1=2B,2=4B,3=8B; dims/strides in elements.
//   dim0 contiguous; dim1 stride = stride0; dim2 (z) stride = stride1.
// LDS fill order (x fastest, then y, then z) is contiguous from lds_addr.
__device__ __forceinline__ TdmDesc tdm_make(
    unsigned lds_addr, const void* gptr, unsigned data_size_log2,
    unsigned tile0, unsigned tile1, unsigned tile2,
    unsigned long long stride0, unsigned long long stride1, unsigned tdim2) {
  TdmDesc d;
  unsigned long long ga = (unsigned long long)(uintptr_t)gptr;
  d.g0[0] = 1u;                                   // count=1, user mode
  d.g0[1] = lds_addr;                             // bits 63:32
  d.g0[2] = (unsigned)ga;                         // global_addr[31:0]
  d.g0[3] = ((unsigned)(ga >> 32) & 0x01ffffffu)  // global_addr[56:32]
            | (2u << 30);                         // type = 2 ("image")
  d.g1[0] = (int)(data_size_log2 << 16);          // data_size; no pad/iter
  d.g1[1] = (int)((tile0 & 0xffffu) << 16);       // tensor_dim0[15:0]
  d.g1[2] = (int)(((tile0 >> 16) & 0xffffu) | ((tile1 & 0xffffu) << 16));
  d.g1[3] = (int)(((tile1 >> 16) & 0xffffu) | ((tile0 & 0xffffu) << 16)); // tile_dim0
  d.g1[4] = (int)((tile1 & 0xffffu) | ((tile2 & 0xffffu) << 16));         // tile_dim1/2
  d.g1[5] = (int)(unsigned)stride0;               // tensor_dim0_stride[31:0]
  d.g1[6] = (int)(((unsigned)(stride0 >> 32) & 0xffffu) |
                  (((unsigned)stride1 & 0xffffu) << 16));
  d.g1[7] = (int)(unsigned)((stride1 >> 16) & 0xffffffffull);
  d.g2[0] = (int)tdim2;                           // tensor_dim2 (z bound)
  d.g2[1] = 0; d.g2[2] = 0; d.g2[3] = 0;
  d.g3[0] = 0; d.g3[1] = 0; d.g3[2] = 0; d.g3[3] = 0;
  return d;
}

#if USE_TDM
__device__ __forceinline__ void tdm_load(const TdmDesc& d) {
#if __clang_major__ >= 23
  i32x8 z = {0, 0, 0, 0, 0, 0, 0, 0};
  __builtin_amdgcn_tensor_load_to_lds(d.g0, d.g1, d.g2, d.g3, z, 0);
#else
  __builtin_amdgcn_tensor_load_to_lds(d.g0, d.g1, d.g2, d.g3, 0);
#endif
}
__device__ __forceinline__ void tdm_store(const TdmDesc& d) {
#if __clang_major__ >= 23
  i32x8 z = {0, 0, 0, 0, 0, 0, 0, 0};
  __builtin_amdgcn_tensor_store_from_lds(d.g0, d.g1, d.g2, d.g3, z, 0);
#else
  __builtin_amdgcn_tensor_store_from_lds(d.g0, d.g1, d.g2, d.g3, 0);
#endif
}
__device__ __forceinline__ void tdm_wait() {
  __builtin_amdgcn_s_wait_tensorcnt(0);
}
#endif

__device__ __forceinline__ unsigned lds_off(const void* p) {
  return (unsigned)(uintptr_t)p;   // generic addr low 32 bits = LDS offset
}

// ---------------------------------------------------------------------------
// Kernel 0: convert weights to f16, fold conv bias into BN scale/shift.
//   y = (dot + b)*s + (beta - mean*s) = dot*s + [beta + (b - mean)*s]
// ---------------------------------------------------------------------------
__global__ __launch_bounds__(256) void prep_kernel(
    const float* __restrict__ w1, const float* __restrict__ b1,
    const float* __restrict__ g1, const float* __restrict__ be1,
    const float* __restrict__ m1, const float* __restrict__ v1,
    const float* __restrict__ w2, const float* __restrict__ b2,
    const float* __restrict__ g2, const float* __restrict__ be2,
    const float* __restrict__ m2, const float* __restrict__ v2,
    _Float16* __restrict__ w1h, _Float16* __restrict__ w2h,
    float* __restrict__ sc1, float* __restrict__ sh1,
    float* __restrict__ sc2, float* __restrict__ sh2) {
  const int i = blockIdx.x * 256 + threadIdx.x;
  if (i < CH_ * C_) {
    w1h[i] = (_Float16)w1[i];
    w2h[i] = (_Float16)w2[i];
  }
  if (i < CH_) {
    float s = g1[i] * rsqrtf(v1[i] + 1e-5f);
    sc1[i] = s;
    sh1[i] = be1[i] + (b1[i] - m1[i]) * s;
  }
  if (i < C_) {
    float s = g2[i] * rsqrtf(v2[i] + 1e-5f);
    sc2[i] = s;
    sh2[i] = be2[i] + (b2[i] - m2[i]) * s;
  }
}

// ---------------------------------------------------------------------------
// Kernel 1: first LIF scan over x [T,B,C,HW] f32 -> spikes1 [T,B,HW,C] f16.
// Coalesced 128B reads along hw; transposed tile staged in LDS and written
// back by one TDM 3D tile store (x=c,y=hw,z=t).
// ---------------------------------------------------------------------------
__global__ __launch_bounds__(256) void lif1_kernel(
    const float* __restrict__ x, _Float16* __restrict__ spk1) {
  __shared__ __align__(16) _Float16 sT[T_][32][64];  // [t][hw][c] 32 KB

  const int c0  = blockIdx.x * 64;
  const int hw0 = blockIdx.y * 32;
  const int b   = blockIdx.z;
  const int tid = threadIdx.x;
  const int hw_l = tid & 31;
  const int cg   = tid >> 5;   // 0..7

#pragma unroll
  for (int cc = 0; cc < 8; ++cc) {
    const int c_l = cg * 8 + cc;
    const float* xp = x + ((size_t)(b * C_ + c0 + c_l)) * HW_ + hw0 + hw_l;
    float v = 0.0f;
#pragma unroll
    for (int t = 0; t < T_; ++t) {
      float xv = xp[(size_t)t * (B_ * C_ * HW_)];
      v += (xv - v) * INV_TAU;              // charge (v_reset = 0)
      float s = (v >= 1.0f) ? 1.0f : 0.0f;  // fire
      v *= (1.0f - s);                      // hard reset
      sT[t][hw_l][c_l] = (_Float16)s;
    }
  }
  __syncthreads();

#if USE_TDM
  if (tid < 32) {
    TdmDesc d = tdm_make(lds_off(&sT[0][0][0]),
                         spk1 + ((size_t)(b * HW_ + hw0)) * C_ + c0,
                         /*f16*/1, /*x=c*/64, /*y=hw*/32, /*z=t*/T_,
                         /*y stride*/C_, /*z stride*/(unsigned long long)B_ * HW_ * C_, T_);
    tdm_store(d);
    tdm_wait();
  }
#else
  const unsigned int* sTu = (const unsigned int*)sT;
  for (int i = tid; i < 8192; i += 256) {
    int row = i >> 5, col = i & 31;
    int t = row >> 5, hw = row & 31;
    *(unsigned int*)(spk1 +
        ((size_t)((t * B_ + b) * HW_ + hw0 + hw)) * C_ + c0 + col * 2) = sTu[i];
  }
#endif
}

// ---------------------------------------------------------------------------
// Kernel 2: GEMM1 (spikes1 x W1^T) + BN + second LIF -> spikes2 [T,B,HW,Ch].
// WG = 8 waves; tile 16 hw (M) x 128 out-ch (N); all 8 t per wave.
// Double-buffered TDM tiles: A = 3D (32k x 16m x 8t), B = 2D (32k x 128o).
// Dynamic LDS: A0|A1 (2x8K), B0|B1 (2x8K), spike stage sS (32K) = 64 KB.
// ---------------------------------------------------------------------------
__global__ __launch_bounds__(256) void gemm1_lif_kernel(
    const _Float16* __restrict__ spk1, const _Float16* __restrict__ w1h,
    const float* __restrict__ sc1, const float* __restrict__ sh1,
    _Float16* __restrict__ spk2) {
  extern __shared__ __align__(16) char smem[];

  const int o0  = blockIdx.x * 128;
  const int hw0 = blockIdx.y * 16;
  const int b   = blockIdx.z;
  const int tid  = threadIdx.x;
  const int lane = tid & 31;
  const int w    = tid >> 5;
  const int row_m = lane & 15;
  const int kb8   = (lane >> 4) * 8;
  const int kb16  = (lane >> 4) * 16;
  const int oL    = w * 16 + (lane & 15);

  const unsigned ldsBase = lds_off(smem);
  const _Float16* gA = spk1 + ((size_t)(b * HW_ + hw0)) * C_;
  const _Float16* gB = w1h + (size_t)o0 * C_;
  const unsigned long long zstr = (unsigned long long)B_ * HW_ * C_;

  v8f acc[T_] = {};

#if USE_TDM
  if (w == 0) {
    tdm_load(tdm_make(ldsBase + 0,     gA, 1, 32, 16,  T_, C_, zstr, T_));
    tdm_load(tdm_make(ldsBase + 16384, gB, 1, 32, 128, 0,  C_, 0,   0));
  }
#endif

  for (int kk = 0, buf = 0; kk < C_; kk += 32, buf ^= 1) {
#if USE_TDM
    if (w == 0) tdm_wait();
    __syncthreads();   // publish buf; all waves done reading buf^1
    if (w == 0 && kk + 32 < C_) {
      int nb = buf ^ 1;
      tdm_load(tdm_make(ldsBase + nb * 8192,         gA + kk + 32,
                        1, 32, 16,  T_, C_, zstr, T_));
      tdm_load(tdm_make(ldsBase + 16384 + nb * 8192, gB + kk + 32,
                        1, 32, 128, 0,  C_, 0,   0));
    }
#else
    __syncthreads();
    for (int i = tid; i < 1024; i += 256) {
      if (i < 512) {
        int r = i >> 2, q = i & 3, t = r >> 4, m = r & 15;
        *(uint4*)(smem + buf * 8192 + (r * 32 + q * 8) * 2) =
            *(const uint4*)(gA + (size_t)t * zstr + (size_t)m * C_ + kk + q * 8);
      } else {
        int j = i - 512, r = j >> 2, q = j & 3;
        *(uint4*)(smem + 16384 + buf * 8192 + (r * 32 + q * 8) * 2) =
            *(const uint4*)(gB + (size_t)r * C_ + kk + q * 8);
      }
    }
    __syncthreads();
#endif
    const _Float16* sAb = (const _Float16*)(smem + buf * 8192);
    const _Float16* sBb = (const _Float16*)(smem + 16384 + buf * 8192);

    v16h bf = *(const v16h*)(sBb + oL * 32 + kb16);
    HFrag af[T_];
#pragma unroll
    for (int t = 0; t < T_; ++t) {   // distinct regs: one drain, 8 WMMAs
      af[t].h[0] = *(const v8h*)(sAb + (t * 16 + row_m) * 32 + kb16);
      af[t].h[1] = *(const v8h*)(sAb + (t * 16 + row_m) * 32 + kb16 + 8);
    }
#pragma unroll
    for (int t = 0; t < T_; ++t)
      acc[t] = __builtin_amdgcn_wmma_f32_16x16x32_f16(
          false, af[t].v, false, bf, (short)0, acc[t], false, false);
  }
  __syncthreads();

  // Epilogue: BN affine + in-register LIF over t, stage spikes in LDS.
  _Float16* sS = (_Float16*)(smem + 32768);   // [t][m][o] 16x128 per t
  const int og = o0 + oL;
  const float sc = sc1[og];
  const float sh = sh1[og];
  const int mBase = (lane < 16) ? 0 : 8;      // C/D layout: hi lanes = M+8
#pragma unroll
  for (int j = 0; j < 8; ++j) {
    float v = 0.0f;
#pragma unroll
    for (int t = 0; t < T_; ++t) {
      float y = acc[t][j] * sc + sh;
      v += (y - v) * INV_TAU;
      float s = (v >= 1.0f) ? 1.0f : 0.0f;
      v *= (1.0f - s);
      sS[(t * 16 + mBase + j) * 128 + oL] = (_Float16)s;
    }
  }
  __syncthreads();

#if USE_TDM
  if (w == 0) {
    TdmDesc d = tdm_make(ldsBase + 32768,
                         spk2 + ((size_t)(b * HW_ + hw0)) * CH_ + o0,
                         1, /*x=ch*/128, /*y=m*/16, /*z=t*/T_,
                         CH_, (unsigned long long)B_ * HW_ * CH_, T_);
    tdm_store(d);
    tdm_wait();
  }
#else
  const unsigned int* sSu = (const unsigned int*)sS;
  for (int i = tid; i < 8192; i += 256) {
    int row = i >> 6, col = i & 63;
    int t = row >> 4, m = row & 15;
    *(unsigned int*)(spk2 +
        ((size_t)((t * B_ + b) * HW_ + hw0 + m)) * CH_ + o0 + col * 2) = sSu[i];
  }
#endif
}

// ---------------------------------------------------------------------------
// Kernel 3: GEMM2 (spikes2 x W2^T) + BN -> out [T,B,C,HW] f32. Per-t
// independent. K_TILE=64 (2 WMMAs per wave per step, 24 steps).
// Dynamic LDS: A0|A1 (2x2K), B0|B1 (2x16K), out stage sO (8K) = 44 KB.
// ---------------------------------------------------------------------------
__global__ __launch_bounds__(256) void gemm2_bn_kernel(
    const _Float16* __restrict__ spk2, const _Float16* __restrict__ w2h,
    const float* __restrict__ sc2, const float* __restrict__ sh2,
    float* __restrict__ out) {
  extern __shared__ __align__(16) char smem[];

  const int o0  = blockIdx.x * 128;
  const int hw0 = blockIdx.y * 16;
  const int tb  = blockIdx.z;   // t*B + b
  const int tid  = threadIdx.x;
  const int lane = tid & 31;
  const int w    = tid >> 5;
  const int row_m = lane & 15;
  const int kb8   = (lane >> 4) * 8;
  const int kb16  = (lane >> 4) * 16;
  const int oL    = w * 16 + (lane & 15);

  const unsigned ldsBase = lds_off(smem);
  const _Float16* gA = spk2 + ((size_t)(tb * HW_ + hw0)) * CH_;
  const _Float16* gB = w2h + (size_t)o0 * CH_;

  v8f acc = {};

#if USE_TDM
  if (w == 0) {
    tdm_load(tdm_make(ldsBase + 0,    gA, 1, 64, 16,  0, CH_, 0, 0));
    tdm_load(tdm_make(ldsBase + 4096, gB, 1, 64, 128, 0, CH_, 0, 0));
  }
#endif

  for (int kk = 0, buf = 0; kk < CH_; kk += 64, buf ^= 1) {
#if USE_TDM
    if (w == 0) tdm_wait();
    __syncthreads();
    if (w == 0 && kk + 64 < CH_) {
      int nb = buf ^ 1;
      tdm_load(tdm_make(ldsBase + nb * 2048,         gA + kk + 64,
                        1, 64, 16,  0, CH_, 0, 0));
      tdm_load(tdm_make(ldsBase + 4096 + nb * 16384, gB + kk + 64,
                        1, 64, 128, 0, CH_, 0, 0));
    }
#else
    __syncthreads();
    for (int i = tid; i < 1280; i += 256) {
      if (i < 256) {
        int r = i >> 3, q = i & 7;
        *(uint4*)(smem + buf * 2048 + (r * 64 + q * 8) * 2) =
            *(const uint4*)(gA + (size_t)r * CH_ + kk + q * 8);
      } else {
        int j = i - 256, r = j >> 3, q = j & 7;
        *(uint4*)(smem + 4096 + buf * 16384 + (r * 64 + q * 8) * 2) =
            *(const uint4*)(gB + (size_t)r * CH_ + kk + q * 8);
      }
    }
    __syncthreads();
#endif
    const _Float16* sAb = (const _Float16*)(smem + buf * 2048);
    const _Float16* sBb = (const _Float16*)(smem + 4096 + buf * 16384);

    HFrag af[2];
    v16h bf[2];
#pragma unroll
    for (int h = 0; h < 2; ++h) {
      bf[h] = *(const v16h*)(sBb + oL * 64 + h * 32 + kb16);
      af[h].h[0] = *(const v8h*)(sAb + row_m * 64 + h * 32 + kb16);
      af[h].h[1] = *(const v8h*)(sAb + row_m * 64 + h * 32 + kb16 + 8);
    }
#pragma unroll
    for (int h = 0; h < 2; ++h)
      acc = __builtin_amdgcn_wmma_f32_16x16x32_f16(
          false, af[h].v, false, bf[h], (short)0, acc, false, false);
  }
  __syncthreads();

  float* sO = (float*)(smem + 36864);   // [o_local][m] 128x16 f32
  const int og = o0 + oL;
  const float sc = sc2[og];
  const float sh = sh2[og];
  const int mBase = (lane < 16) ? 0 : 8;
#pragma unroll
  for (int j = 0; j < 8; ++j)
    sO[oL * 16 + mBase + j] = acc[j] * sc + sh;
  __syncthreads();

#if USE_TDM
  if (w == 0) {
    TdmDesc d = tdm_make(ldsBase + 36864,
                         out + ((size_t)tb * C_ + o0) * HW_ + hw0,
                         /*f32*/2, /*x=hw*/16, /*y=o*/128, 0, HW_, 0, 0);
    tdm_store(d);
    tdm_wait();
  }
#else
  const float* sOf = (const float*)sO;
  for (int i = tid; i < 2048; i += 256) {
    int r = i >> 4, m = i & 15;
    out[((size_t)tb * C_ + o0 + r) * HW_ + hw0 + m] = sOf[i];
  }
#endif
}

// ---------------------------------------------------------------------------
extern "C" void kernel_launch(void* const* d_in, const int* in_sizes, int n_in,
                              void* d_out, int out_size, void* d_ws, size_t ws_size,
                              hipStream_t stream) {
  (void)in_sizes; (void)n_in; (void)out_size; (void)ws_size;

  const float* x   = (const float*)d_in[0];
  const float* w1  = (const float*)d_in[1];
  const float* b1  = (const float*)d_in[2];
  const float* g1  = (const float*)d_in[3];
  const float* be1 = (const float*)d_in[4];
  const float* m1  = (const float*)d_in[5];
  const float* v1  = (const float*)d_in[6];
  const float* w2  = (const float*)d_in[7];
  const float* b2  = (const float*)d_in[8];
  const float* g2  = (const float*)d_in[9];
  const float* be2 = (const float*)d_in[10];
  const float* m2  = (const float*)d_in[11];
  const float* v2  = (const float*)d_in[12];
  float* out = (float*)d_out;

  char* ws = (char*)d_ws;
  size_t off = 0;
  auto alloc = [&](size_t bytes) -> void* {
    void* p = ws + off;
    off = (off + bytes + 255) & ~(size_t)255;
    return p;
  };

  _Float16* w1h  = (_Float16*)alloc((size_t)CH_ * C_ * 2);
  _Float16* w2h  = (_Float16*)alloc((size_t)C_ * CH_ * 2);
  float*    sc1  = (float*)alloc(CH_ * 4);
  float*    sh1  = (float*)alloc(CH_ * 4);
  float*    sc2  = (float*)alloc(C_ * 4);
  float*    sh2  = (float*)alloc(C_ * 4);
  _Float16* spk1 = (_Float16*)alloc((size_t)T_ * B_ * HW_ * C_ * 2);   // ~50 MB
  _Float16* spk2 = (_Float16*)alloc((size_t)T_ * B_ * HW_ * CH_ * 2);  // ~201 MB

  prep_kernel<<<(CH_ * C_ + 255) / 256, 256, 0, stream>>>(
      w1, b1, g1, be1, m1, v1, w2, b2, g2, be2, m2, v2,
      w1h, w2h, sc1, sh1, sc2, sh2);

  lif1_kernel<<<dim3(C_ / 64, HW_ / 32, B_), 256, 0, stream>>>(x, spk1);

  gemm1_lif_kernel<<<dim3(CH_ / 128, HW_ / 16, B_), 256, 65536, stream>>>(
      spk1, w1h, sc1, sh1, spk2);

  gemm2_bn_kernel<<<dim3(C_ / 128, HW_ / 16, T_ * B_), 256, 45056, stream>>>(
      spk2, w2h, sc2, sh2, out);
}
